// NeuralAttention_45956150067931
// MI455X (gfx1250) — matrix operand order repaired and checked by reading, not compiled, not run on hardware
//
#include <hip/hip_runtime.h>
#include <hip/hip_bf16.h>
#include <math.h>
#include <stdint.h>

#define B_      4
#define L_      512
#define T_      4096
#define HID_    1024
#define NH_     16
#define HD_     64
#define MAXPOS_ 4096

typedef __attribute__((ext_vector_type(16))) _Float16     v16h;
typedef __attribute__((ext_vector_type(8)))  float        v8f;
typedef __attribute__((ext_vector_type(4)))  unsigned int v4u;

union Frag { v16h h16; v4u u[2]; };

// ---------------------------------------------------------------------------
// Tensor Data Mover availability (5-arg on ROCm 7.2, 6-arg on amdgpu-toolchain
// whose therock headers are the only place amd_gfx1250_TDM.h exists).
// ---------------------------------------------------------------------------
#if defined(__HIP_DEVICE_COMPILE__) && __has_builtin(__builtin_amdgcn_tensor_load_to_lds)
#define HAVE_TDM 1
#else
#define HAVE_TDM 0
#endif

#if __has_include(<hip/amd_detail/amd_gfx1250_TDM.h>)
#define TDM_6ARG 1
#else
#define TDM_6ARG 0
#endif

#if HAVE_TDM
typedef __attribute__((ext_vector_type(4))) unsigned int u32x4;
typedef __attribute__((ext_vector_type(8))) int          i32x8;
typedef __attribute__((ext_vector_type(4))) int          i32x4;

// 2-D f16 tile load Global->LDS via TDM. Descriptor packing per CDNA5 ISA
// (CDNA5 ISA): D# group0 = {count=1, lds_addr,
// global_addr[56:0], type=2}; group1 = {wg_mask=0, data_size=1 (2B),
// tensor_dim0/1, tile_dim0/1, tensor_dim0_stride}. Groups 2/3 zero (<=2D).
static __device__ inline void tdm_load_tile_2d(unsigned lds_addr, const void* gptr,
                                               unsigned tile_d0, unsigned tile_d1,
                                               unsigned long long row_stride_elems) {
    unsigned long long ga = (unsigned long long)(uintptr_t)gptr;
    u32x4 g0;
    g0[0] = 1u;                                                   // count=1, user D#
    g0[1] = lds_addr;                                             // LDS byte addr
    g0[2] = (unsigned)(ga & 0xffffffffull);                       // global_addr[31:0]
    g0[3] = (unsigned)((ga >> 32) & 0x01ffffffull) | 0x80000000u; // [56:32] | type=2
    i32x8 g1;
    g1[0] = 0x00010000;                                           // data_size=1 (2B)
    g1[1] = (int)((tile_d0 & 0xffffu) << 16);                     // tensor_dim0 lo
    g1[2] = (int)(((tile_d0 >> 16) & 0xffffu) | ((tile_d1 & 0xffffu) << 16));
    g1[3] = (int)(((tile_d1 >> 16) & 0xffffu) | ((tile_d0 & 0xffffu) << 16)); // tile_dim0
    g1[4] = (int)(tile_d1 & 0xffffu);                             // tile_dim1 (dim2=0)
    g1[5] = (int)(row_stride_elems & 0xffffffffull);              // dim0_stride lo
    g1[6] = (int)((row_stride_elems >> 32) & 0xffffull);          // dim0_stride hi
    g1[7] = 0;
    i32x4 gz = {0, 0, 0, 0};
#if TDM_6ARG
    i32x8 gz8 = {0, 0, 0, 0, 0, 0, 0, 0};
    __builtin_amdgcn_tensor_load_to_lds(g0, g1, gz, gz, gz8, 0);
#else
    __builtin_amdgcn_tensor_load_to_lds(g0, g1, gz, gz, 0);
#endif
}
#endif // HAVE_TDM

// ---------------------------------------------------------------------------
// WMMA helpers: D = A(16x32 f16) x B(32x16 f16) + C(16x16 f32)
// ---------------------------------------------------------------------------
static __device__ inline v8f wmma_f16(v16h a, v16h b, v8f c) {
    return __builtin_amdgcn_wmma_f32_16x16x32_f16(
        /*neg_a=*/false, a, /*neg_b=*/false, b,
        /*c_mod=*/(short)0, c, /*reuse_a=*/false, /*reuse_b=*/false);
}

// A-matrix 16x32 fragment: lane l holds row (l&15); halves at
// k = k0 + 8*hi .. +7 and k = k0 + 16 + 8*hi .. +7  (hi = l>>4).
static __device__ inline v16h load_a_frag(const _Float16* base, int ld, int row, int k0) {
    int lane = threadIdx.x & 31;
    int r    = row + (lane & 15);
    int kb   = (lane >> 4) * 8;
    const _Float16* p = base + (size_t)r * ld + k0 + kb;
    Frag f;
    f.u[0] = *(const v4u*)(p);
    f.u[1] = *(const v4u*)(p + 16);
    return f.h16;
}

// B-matrix 32x16 fragment from an n-major source: lane l holds column
// n = col + (l&15); halves at k = k0 + 8*hi .. +7 and k = k0 + 16 + 8*hi .. +7.
static __device__ inline v16h load_b_frag(const _Float16* base, int ld, int col, int k0) {
    int lane = threadIdx.x & 31;
    int n    = col + (lane & 15);
    int kb   = (lane >> 4) * 8;
    const _Float16* p = base + (size_t)n * ld + k0 + kb;
    Frag f;
    f.u[0] = *(const v4u*)(p);
    f.u[1] = *(const v4u*)(p + 16);
    return f.h16;
}

// ---------------------------------------------------------------------------
// fp32 -> fp16 convert
// ---------------------------------------------------------------------------
__global__ void f2h_kernel(const float* __restrict__ in, _Float16* __restrict__ out, size_t n) {
    size_t i = (size_t)blockIdx.x * blockDim.x + threadIdx.x;
    if (i < n) out[i] = (_Float16)in[i];
}

// ---------------------------------------------------------------------------
// GEMM: out[m][n] = sum_k A16[m][k] * W16[n][k] + bias[n]
// Block = 128 threads (4 waves). Block tile 64x64; wave tile 16x64.
// ---------------------------------------------------------------------------
template <bool OUT_F32>
__global__ void gemm_nt_kernel(const _Float16* __restrict__ A,
                               const _Float16* __restrict__ W,
                               const float* __restrict__ bias,
                               float* __restrict__ outF,
                               _Float16* __restrict__ outH,
                               int M, int N, int K) {
    int wave = threadIdx.x >> 5;
    int lane = threadIdx.x & 31;
    int m0   = (blockIdx.y * 4 + wave) * 16;
    int n0   = blockIdx.x * 64;

    v8f acc[4] = {};
    for (int k0 = 0; k0 < K; k0 += 32) {
        v16h a = load_a_frag(A, K, m0, k0);
        if (k0 + 32 < K) // pull next A chunk toward the WGP (global_prefetch_b8)
            __builtin_prefetch(A + (size_t)(m0 + (lane & 15)) * K + k0 + 32, 0, 3);
#pragma unroll
        for (int j = 0; j < 4; ++j) {
            v16h b = load_b_frag(W, K, n0 + j * 16, k0);
            acc[j] = wmma_f16(a, b, acc[j]);
        }
    }

    int hi = lane >> 4;
    int nc = lane & 15;
#pragma unroll
    for (int j = 0; j < 4; ++j) {
        int   n  = n0 + j * 16 + nc;
        float bv = bias ? bias[n] : 0.0f;
#pragma unroll
        for (int i = 0; i < 8; ++i) {
            int   m = m0 + i + 8 * hi;
            float v = acc[j][i] + bv;
            if (OUT_F32) outF[(size_t)m * N + n] = v;
            else         outH[(size_t)m * N + n] = (_Float16)v;
        }
    }
}

// ---------------------------------------------------------------------------
// RoPE + repack [B*seq, HID] f16 -> [B, NH, seq, HD] f16 (optionally scaled).
// ---------------------------------------------------------------------------
__global__ void rope_pack_kernel(const _Float16* __restrict__ in,
                                 _Float16* __restrict__ out,
                                 const int* __restrict__ ts,
                                 int seq, float scale) {
    size_t idx   = (size_t)blockIdx.x * blockDim.x + threadIdx.x;
    size_t total = (size_t)B_ * seq * NH_ * 32;
    if (idx >= total) return;
    int    j  = (int)(idx & 31);
    size_t t1 = idx >> 5;
    int    h  = (int)(t1 % NH_);
    size_t t2 = t1 / NH_;
    int    i  = (int)(t2 % seq);
    int    b  = (int)(t2 / seq);

    int pos = ts ? ts[(size_t)b * seq + i]
                 : (int)(((double)i * (double)(MAXPOS_ - 1)) / (double)(L_ - 1));

    float invf = expf(-0.28782313662425572f * (float)j); // ln(10000)/32
    float ang  = (float)pos * invf;
    float c = cosf(ang), s = sinf(ang);

    const _Float16* src = in + ((size_t)b * seq + i) * HID_ + h * HD_;
    float x1 = (float)src[j];
    float x2 = (float)src[j + 32];

    _Float16* dst = out + (((size_t)(b * NH_ + h)) * seq + i) * HD_;
    dst[j]      = (_Float16)((x1 * c - x2 * s) * scale);
    dst[j + 32] = (_Float16)((x2 * c + x1 * s) * scale);
}

// ---------------------------------------------------------------------------
// V repack: [B*T, HID] f16 -> Vt [B, NH, HD, T] f16.
// ---------------------------------------------------------------------------
__global__ void vpack_kernel(const _Float16* __restrict__ in, _Float16* __restrict__ out) {
    size_t idx   = (size_t)blockIdx.x * blockDim.x + threadIdx.x;
    size_t total = (size_t)B_ * T_ * HID_;
    if (idx >= total) return;
    int    c  = (int)(idx % HID_);
    size_t t1 = idx / HID_;
    int    t  = (int)(t1 % T_);
    int    b  = (int)(t1 / T_);
    int    h  = c / HD_, d = c % HD_;
    out[(((size_t)(b * NH_ + h)) * HD_ + d) * T_ + t] = in[idx];
}

// ---------------------------------------------------------------------------
// Flash attention. grid = (L/64, B*NH), block = 128 (4 waves); all 4 waves of
// a block share (b,h), so K/V 32-column tiles are staged ONCE per block into
// double-buffered LDS by the Tensor Data Mover (wave 0 issues, tracked with
// TENSORcnt, s_wait_tensorcnt + workgroup barrier), overlapping the DMA of
// chunk i+1 with the WMMA/softmax of chunk i.
// ---------------------------------------------------------------------------
__global__ void attn_kernel(const _Float16* __restrict__ q16,
                            const _Float16* __restrict__ k16,
                            const _Float16* __restrict__ vt16,
                            const unsigned char* __restrict__ mask,
                            _Float16* __restrict__ o16) {
    __shared__ _Float16 kt[2][32][64];   // K chunk:  rows t (32) x d (64)
    __shared__ _Float16 vv[2][64][32];   // Vt chunk: rows d (64) x t (32)
    __shared__ _Float16 plds[4][16][40]; // per-wave P staging (padded rows)

    int wave = threadIdx.x >> 5;
    int lane = threadIdx.x & 31;
    int hi   = lane >> 4;
    int nc   = lane & 15;

    int bh = blockIdx.y;
    int b  = bh / NH_;
    int h  = bh % NH_;
    int l0 = (blockIdx.x * 4 + wave) * 16;

    const _Float16* Qbh = q16  + (size_t)bh * L_ * HD_;
    const _Float16* Kbh = k16  + (size_t)bh * T_ * HD_;
    const _Float16* Vbh = vt16 + (size_t)bh * HD_ * T_;
    const unsigned char* mb = mask + (size_t)b * T_;

    v16h qa0 = load_a_frag(Qbh, HD_, l0, 0);
    v16h qa1 = load_a_frag(Qbh, HD_, l0, 32);

    v8f   o[4] = {};
    float rm[8], rs[8];
#pragma unroll
    for (int i = 0; i < 8; ++i) { rm[i] = -1e30f; rs[i] = 0.0f; }

    const int NIT = T_ / 32;
    int buf = 0;

#if HAVE_TDM
    if (wave == 0) {
        tdm_load_tile_2d((unsigned)(uintptr_t)&kt[0][0][0], Kbh, 64, 32, 64);
        tdm_load_tile_2d((unsigned)(uintptr_t)&vv[0][0][0], Vbh, 32, 64, (unsigned long long)T_);
    }
#endif

    for (int it = 0; it < NIT; ++it) {
        int t0 = it * 32;
        v8f slo = {}, shi = {};

#if HAVE_TDM
        if (wave == 0) {
            if (it + 1 < NIT) { // prefetch next chunk into the other buffer
                int tn = t0 + 32;
                tdm_load_tile_2d((unsigned)(uintptr_t)&kt[buf ^ 1][0][0],
                                 Kbh + (size_t)tn * HD_, 64, 32, 64);
                tdm_load_tile_2d((unsigned)(uintptr_t)&vv[buf ^ 1][0][0],
                                 Vbh + tn, 32, 64, (unsigned long long)T_);
                __builtin_amdgcn_s_wait_tensorcnt(2); // this chunk's pair landed
            } else {
                __builtin_amdgcn_s_wait_tensorcnt(0);
            }
        }
        __syncthreads(); // LDS tiles visible to all 4 waves

        const _Float16* kb = &kt[buf][0][0];
        slo = wmma_f16(qa0, load_b_frag(kb, 64, 0,  0),  slo);
        slo = wmma_f16(qa1, load_b_frag(kb, 64, 0,  32), slo);
        shi = wmma_f16(qa0, load_b_frag(kb, 64, 16, 0),  shi);
        shi = wmma_f16(qa1, load_b_frag(kb, 64, 16, 32), shi);
#else
        slo = wmma_f16(qa0, load_b_frag(Kbh, HD_, t0,      0),  slo);
        slo = wmma_f16(qa1, load_b_frag(Kbh, HD_, t0,      32), slo);
        shi = wmma_f16(qa0, load_b_frag(Kbh, HD_, t0 + 16, 0),  shi);
        shi = wmma_f16(qa1, load_b_frag(Kbh, HD_, t0 + 16, 32), shi);
#endif

        // ---- mask ----
        bool mlo = mb[t0 + nc]      != 0;
        bool mhi = mb[t0 + 16 + nc] != 0;
        float sl[8], sh[8];
#pragma unroll
        for (int i = 0; i < 8; ++i) {
            sl[i] = mlo ? slo[i] : -1e30f;
            sh[i] = mhi ? shi[i] : -1e30f;
        }

        // ---- online softmax ----
        float cm[8];
#pragma unroll
        for (int i = 0; i < 8; ++i) cm[i] = fmaxf(sl[i], sh[i]);
#pragma unroll
        for (int off = 1; off < 16; off <<= 1)
#pragma unroll
            for (int i = 0; i < 8; ++i)
                cm[i] = fmaxf(cm[i], __shfl_xor(cm[i], off, 32));

        float sf[8], ps[8];
#pragma unroll
        for (int i = 0; i < 8; ++i) {
            float mn = fmaxf(rm[i], cm[i]);
            sf[i]    = expf(rm[i] - mn);
            rm[i]    = mn;
            float pl = expf(sl[i] - mn);
            float ph = expf(sh[i] - mn);
            ps[i]    = pl + ph;
            plds[wave][i + 8 * hi][nc]      = (_Float16)pl;
            plds[wave][i + 8 * hi][nc + 16] = (_Float16)ph;
        }
#pragma unroll
        for (int off = 1; off < 16; off <<= 1)
#pragma unroll
            for (int i = 0; i < 8; ++i)
                ps[i] += __shfl_xor(ps[i], off, 32);
#pragma unroll
        for (int i = 0; i < 8; ++i) rs[i] = rs[i] * sf[i] + ps[i];
#pragma unroll
        for (int j = 0; j < 4; ++j)
#pragma unroll
            for (int i = 0; i < 8; ++i) o[j][i] *= sf[i];

        // ---- P: C-layout -> A-layout via LDS ----
        const _Float16* pr = &plds[wave][nc][8 * hi];
        Frag pf;
        pf.u[0] = *(const v4u*)(pr);
        pf.u[1] = *(const v4u*)(pr + 16);
        v16h pa = pf.h16;

        // ---- O += P * V ----
#if HAVE_TDM
        const _Float16* vb = &vv[buf][0][0];
#pragma unroll
        for (int j = 0; j < 4; ++j)
            o[j] = wmma_f16(pa, load_b_frag(vb, 32, j * 16, 0), o[j]);
        __syncthreads(); // all waves done with this buffer before TDM overwrites
        buf ^= 1;
#else
#pragma unroll
        for (int j = 0; j < 4; ++j)
            o[j] = wmma_f16(pa, load_b_frag(Vbh, T_, j * 16, t0), o[j]);
#endif
    }

    // ---- normalize and store [B, L, HID] f16 ----
    float rinv[8];
#pragma unroll
    for (int i = 0; i < 8; ++i) rinv[i] = 1.0f / rs[i];
#pragma unroll
    for (int j = 0; j < 4; ++j)
#pragma unroll
        for (int i = 0; i < 8; ++i) {
            int l = l0 + i + 8 * hi;
            int d = j * 16 + nc;
            o16[((size_t)(b * L_ + l)) * HID_ + h * HD_ + d] =
                (_Float16)(o[j][i] * rinv[i]);
        }
}

// ---------------------------------------------------------------------------
// Host: orchestrate the pipeline on `stream`.
// ---------------------------------------------------------------------------
extern "C" void kernel_launch(void* const* d_in, const int* in_sizes, int n_in,
                              void* d_out, int out_size, void* d_ws, size_t ws_size,
                              hipStream_t stream) {
    (void)in_sizes; (void)n_in; (void)out_size; (void)ws_size;

    const float*         latents = (const float*)d_in[0];
    const float*         target  = (const float*)d_in[1];
    const unsigned char* mask    = (const unsigned char*)d_in[2];
    const int*           ts      = (const int*)d_in[3];
    const float* Wq = (const float*)d_in[4];
    const float* bq = (const float*)d_in[5];
    const float* Wk = (const float*)d_in[6];
    const float* bk = (const float*)d_in[7];
    const float* Wv = (const float*)d_in[8];
    const float* bv = (const float*)d_in[9];
    const float* Wo = (const float*)d_in[10];
    const float* bo = (const float*)d_in[11];
    float* out = (float*)d_out;

    char*  ws  = (char*)d_ws;
    size_t off = 0;
    auto alloc = [&](size_t bytes) -> char* {
        char* p = ws + off;
        off += (bytes + 255) & ~(size_t)255;
        return p;
    };

    const size_t ML = (size_t)B_ * L_;  // 2048 query rows
    const size_t MT = (size_t)B_ * T_;  // 16384 kv rows

    _Float16* tmp16 = (_Float16*)alloc(MT * HID_ * 2);          // reused Q/K/V GEMM out
    _Float16* lat16 = (_Float16*)alloc(ML * HID_ * 2);
    _Float16* tgt16 = (_Float16*)alloc(MT * HID_ * 2);
    _Float16* wq16  = (_Float16*)alloc((size_t)HID_ * HID_ * 2);
    _Float16* wk16  = (_Float16*)alloc((size_t)HID_ * HID_ * 2);
    _Float16* wv16  = (_Float16*)alloc((size_t)HID_ * HID_ * 2);
    _Float16* wo16  = (_Float16*)alloc((size_t)HID_ * HID_ * 2);
    _Float16* q16   = (_Float16*)alloc(ML * HID_ * 2);          // [B,NH,L,HD]
    _Float16* k16   = (_Float16*)alloc(MT * HID_ * 2);          // [B,NH,T,HD]
    _Float16* vt16  = (_Float16*)alloc(MT * HID_ * 2);          // [B,NH,HD,T]
    _Float16* o16   = (_Float16*)alloc(ML * HID_ * 2);          // [B,L,HID]

    auto cvt = [&](const float* src, _Float16* dst, size_t n) {
        f2h_kernel<<<(unsigned)((n + 255) / 256), 256, 0, stream>>>(src, dst, n);
    };
    cvt(latents, lat16, ML * HID_);
    cvt(target,  tgt16, MT * HID_);
    cvt(Wq, wq16, (size_t)HID_ * HID_);
    cvt(Wk, wk16, (size_t)HID_ * HID_);
    cvt(Wv, wv16, (size_t)HID_ * HID_);
    cvt(Wo, wo16, (size_t)HID_ * HID_);

    const float qscale = 0.125f; // 1/sqrt(HD)
    dim3 gemmBlk(128);

    // Q = latents @ Wq^T + bq ; RoPE(latent positions), scaled by 1/sqrt(HD)
    gemm_nt_kernel<false><<<dim3(HID_ / 64, (unsigned)(ML / 64)), gemmBlk, 0, stream>>>(
        lat16, wq16, bq, nullptr, tmp16, (int)ML, HID_, HID_);
    {
        size_t n = (size_t)B_ * L_ * NH_ * 32;
        rope_pack_kernel<<<(unsigned)((n + 255) / 256), 256, 0, stream>>>(
            tmp16, q16, nullptr, L_, qscale);
    }

    // K = target @ Wk^T + bk ; RoPE(target_timestamp)
    gemm_nt_kernel<false><<<dim3(HID_ / 64, (unsigned)(MT / 64)), gemmBlk, 0, stream>>>(
        tgt16, wk16, bk, nullptr, tmp16, (int)MT, HID_, HID_);
    {
        size_t n = (size_t)B_ * T_ * NH_ * 32;
        rope_pack_kernel<<<(unsigned)((n + 255) / 256), 256, 0, stream>>>(
            tmp16, k16, ts, T_, 1.0f);
    }

    // V = target @ Wv^T + bv ; transpose-pack per head
    gemm_nt_kernel<false><<<dim3(HID_ / 64, (unsigned)(MT / 64)), gemmBlk, 0, stream>>>(
        tgt16, wv16, bv, nullptr, tmp16, (int)MT, HID_, HID_);
    {
        size_t n = MT * HID_;
        vpack_kernel<<<(unsigned)((n + 255) / 256), 256, 0, stream>>>(tmp16, vt16);
    }

    // Flash attention (TDM-staged K/V tiles)
    attn_kernel<<<dim3(L_ / 64, B_ * NH_), dim3(128), 0, stream>>>(
        q16, k16, vt16, mask, o16);

    // out = O @ Wo^T + bo (fp32)
    gemm_nt_kernel<true><<<dim3(HID_ / 64, (unsigned)(ML / 64)), gemmBlk, 0, stream>>>(
        o16, wo16, bo, out, nullptr, (int)ML, HID_, HID_);
}
